// SwinTransformerBlock3D_5282809774675
// MI455X (gfx1250) — matrix operand order repaired and checked by reading, not compiled, not run on hardware
//
#include <hip/hip_runtime.h>
#include <hip/hip_fp16.h>

// ---------------------------------------------------------------------------
// Swin-Transformer-3D block for gfx1250 (MI455X), WMMA f16 (f32 accum) GEMMs.
// wave32; v_wmma_f32_16x16x32_f16 everywhere; LDS staging in attention.
// ---------------------------------------------------------------------------

typedef __attribute__((ext_vector_type(16))) _Float16 v16h;
typedef __attribute__((ext_vector_type(8)))  float    v8f;

__device__ __forceinline__ v8f wmma_f16(v16h a, v16h b, v8f c) {
  // (neg_a, A, neg_b, B, c_mod, C, reuse_a, reuse_b)
  return __builtin_amdgcn_wmma_f32_16x16x32_f16(false, a, false, b, (short)0, c, false, false);
}

// A-frag: 16x32 (MxK) f16, row-major source (K contiguous), contiguous k order, matching load_b_frag:
// lane = hf*16+r -> row r, K in {hf*16 .. hf*16+15}
__device__ __forceinline__ v16h load_a_frag(const _Float16* __restrict__ src, int lda) {
  const int lane = threadIdx.x & 31;
  const int r = lane & 15, hf = lane >> 4;
  const _Float16* p = src + (size_t)r * lda + hf * 16;
  v16h a;
#pragma unroll
  for (int j = 0; j < 16; ++j) a[j] = p[j];
  return a;
}

// B-frag: 32x16 (KxN) f16 where B[k][n] = W[n][k], W row-major (K contiguous):
// lane = kh*16+n -> column n, K in {kh*16 .. kh*16+15} (one 32B contiguous chunk)
__device__ __forceinline__ v16h load_b_frag(const _Float16* __restrict__ w, int ldw) {
  const int lane = threadIdx.x & 31;
  const int n = lane & 15, kh = lane >> 4;
  const _Float16* p = w + (size_t)n * ldw + kh * 16;
  v16h b;
#pragma unroll
  for (int j = 0; j < 16; ++j) b[j] = p[j];
  return b;
}

// One wave computes a 16x64 strip: 4 f32 accumulators, K-loop fully unrolled.
template <int K>
__device__ __forceinline__ void wave_gemm(const _Float16* __restrict__ A, int lda,
                                          const _Float16* __restrict__ W, int ldw,
                                          v8f acc[4]) {
#pragma unroll
  for (int k = 0; k < K; k += 32) {
    v16h a = load_a_frag(A + k, lda);
#pragma unroll
    for (int t = 0; t < 4; ++t) {
      v16h b = load_b_frag(W + (size_t)(t * 16) * ldw + k, ldw);
      acc[t] = wmma_f16(a, b, acc[t]);
    }
  }
}

// ---------------------------------------------------------------------------
// small utility kernels
// ---------------------------------------------------------------------------
__global__ void cvt_f16_kernel(const float* __restrict__ s, _Float16* __restrict__ d, int n) {
  int i = blockIdx.x * 256 + threadIdx.x;
  if (i < n) d[i] = (_Float16)s[i];
}

// Relative-position bias gather: biasb[h][n][m], n,m in [0,98), 6 heads.
__global__ void bias_gather_kernel(const float* __restrict__ tbl, float* __restrict__ biasb) {
  int i = blockIdx.x * 256 + threadIdx.x;
  if (i >= 98 * 98) return;
  int n = i / 98, m = i - n * 98;
  int zn = n / 49, yn = (n % 49) / 7, xn = n % 7;
  int zm = m / 49, ym = (m % 49) / 7, xm = m % 7;
  int idx = (zn - zm + 1) * 169 + (yn - ym + 6) * 13 + (xn - xm + 6);
#pragma unroll
  for (int h = 0; h < 6; ++h)
    biasb[(size_t)h * 9604 + i] = tbl[idx * 6 + h];
}

// ---------------------------------------------------------------------------
// LN1 + cyclic shift + window partition -> Xw f16 [1024 windows][112 rows][192]
// 4 tokens per 128-thread block (1 wave per token; wave32 reduction over C=192)
// ---------------------------------------------------------------------------
__global__ void __launch_bounds__(128)
ln1_shift_part(const float* __restrict__ x, const float* __restrict__ g,
               const float* __restrict__ bb, _Float16* __restrict__ xw) {
  const int token = blockIdx.x * 4 + (threadIdx.x >> 5);
  const int lane  = threadIdx.x & 31;
  const int b_ = token / 112, n = token - b_ * 112;
  _Float16* outp = xw + (size_t)token * 192;
  if (n >= 98) {  // zero pad rows (wave-uniform branch)
    for (int j = lane; j < 192; j += 32) outp[j] = (_Float16)0.0f;
    return;
  }
  const int batch = b_ >> 8, wid = b_ & 255;
  const int dw = wid >> 6, hw = (wid >> 3) & 7, ww = wid & 7;
  const int nd = n / 49, rr = n - nd * 49, nh = rr / 7, nw = rr - nh * 7;
  const int d = (dw * 2 + nd + 1) & 7;
  int h = hw * 7 + nh + 3; if (h >= 56) h -= 56;
  int w = ww * 7 + nw + 3; if (w >= 56) w -= 56;
  const float* src = x + ((((size_t)batch * 8 + d) * 56 + h) * 56 + w) * 192;

  float v[6]; float s = 0.f;
#pragma unroll
  for (int j = 0; j < 6; ++j) { v[j] = src[lane + 32 * j]; s += v[j]; }
#pragma unroll
  for (int m = 16; m; m >>= 1) s += __shfl_xor(s, m, 32);
  const float mean = s * (1.0f / 192.0f);
  float q = 0.f;
#pragma unroll
  for (int j = 0; j < 6; ++j) { float dd = v[j] - mean; q += dd * dd; }
#pragma unroll
  for (int m = 16; m; m >>= 1) q += __shfl_xor(q, m, 32);
  const float rstd = rsqrtf(q * (1.0f / 192.0f) + 1e-5f);
#pragma unroll
  for (int j = 0; j < 6; ++j) {
    int c = lane + 32 * j;
    outp[c] = (_Float16)((v[j] - mean) * rstd * g[c] + bb[c]);
  }
}

// LN2 on x2 (natural token order) -> h2 f16
__global__ void __launch_bounds__(128)
ln2_kernel(const float* __restrict__ x2, const float* __restrict__ g,
           const float* __restrict__ bb, _Float16* __restrict__ h2) {
  const int token = blockIdx.x * 4 + (threadIdx.x >> 5);
  const int lane  = threadIdx.x & 31;
  const float* src = x2 + (size_t)token * 192;
  _Float16* outp = h2 + (size_t)token * 192;
  float v[6]; float s = 0.f;
#pragma unroll
  for (int j = 0; j < 6; ++j) { v[j] = src[lane + 32 * j]; s += v[j]; }
#pragma unroll
  for (int m = 16; m; m >>= 1) s += __shfl_xor(s, m, 32);
  const float mean = s * (1.0f / 192.0f);
  float q = 0.f;
#pragma unroll
  for (int j = 0; j < 6; ++j) { float dd = v[j] - mean; q += dd * dd; }
#pragma unroll
  for (int m = 16; m; m >>= 1) q += __shfl_xor(q, m, 32);
  const float rstd = rsqrtf(q * (1.0f / 192.0f) + 1e-5f);
#pragma unroll
  for (int j = 0; j < 6; ++j) {
    int c = lane + 32 * j;
    outp[c] = (_Float16)((v[j] - mean) * rstd * g[c] + bb[c]);
  }
}

// ---------------------------------------------------------------------------
// GEMM kernels: 256 threads = 8 waves; block tile 128x64; wave tile 16x64.
// ---------------------------------------------------------------------------
__global__ void __launch_bounds__(256)
qkv_gemm(const _Float16* __restrict__ A, const _Float16* __restrict__ W,
         const float* __restrict__ bias, _Float16* __restrict__ out) {
  const int wave = threadIdx.x >> 5;
  const int m0 = blockIdx.x * 128 + wave * 16;
  const int n0 = blockIdx.y * 64;
  v8f acc[4] = {};
  wave_gemm<192>(A + (size_t)m0 * 192, 192, W + (size_t)n0 * 192, 192, acc);
  const int lane = threadIdx.x & 31, nn = lane & 15, mh = lane >> 4;
#pragma unroll
  for (int t = 0; t < 4; ++t) {
    const int col = n0 + t * 16 + nn;
    const float bv = bias[col];
#pragma unroll
    for (int r = 0; r < 8; ++r)
      out[(size_t)(m0 + mh * 8 + r) * 576 + col] = (_Float16)(acc[t][r] + bv);
  }
}

// proj GEMM + window-reverse + reverse cyclic shift + residual -> x2 (f32)
__global__ void __launch_bounds__(256)
proj_gemm(const _Float16* __restrict__ A, const _Float16* __restrict__ W,
          const float* __restrict__ bias, const float* __restrict__ shortcut,
          float* __restrict__ x2) {
  const int wave = threadIdx.x >> 5;
  const int m0 = blockIdx.x * 128 + wave * 16;
  const int n0 = blockIdx.y * 64;
  v8f acc[4] = {};
  wave_gemm<192>(A + (size_t)m0 * 192, 192, W + (size_t)n0 * 192, 192, acc);
  const int lane = threadIdx.x & 31, nn = lane & 15, mh = lane >> 4;
#pragma unroll
  for (int t = 0; t < 4; ++t) {
    const int col = n0 + t * 16 + nn;
    const float bv = bias[col];
#pragma unroll
    for (int r = 0; r < 8; ++r) {
      const int row = m0 + mh * 8 + r;
      const int b_ = row / 112, n = row - b_ * 112;
      if (n < 98) {
        const int batch = b_ >> 8, wid = b_ & 255;
        const int dw = wid >> 6, hw = (wid >> 3) & 7, ww = wid & 7;
        const int nd = n / 49, rr = n - nd * 49, nh = rr / 7, nw = rr - nh * 7;
        const int d = (dw * 2 + nd + 1) & 7;
        int h = hw * 7 + nh + 3; if (h >= 56) h -= 56;
        int w = ww * 7 + nw + 3; if (w >= 56) w -= 56;
        const size_t idx = ((((size_t)batch * 8 + d) * 56 + h) * 56 + w) * 192 + col;
        x2[idx] = shortcut[idx] + acc[t][r] + bv;
      }
    }
  }
}

// fc1 GEMM + exact GELU -> f16
__global__ void __launch_bounds__(256)
fc1_gemm(const _Float16* __restrict__ A, const _Float16* __restrict__ W,
         const float* __restrict__ bias, _Float16* __restrict__ out) {
  const int wave = threadIdx.x >> 5;
  const int m0 = blockIdx.x * 128 + wave * 16;
  const int n0 = blockIdx.y * 64;
  v8f acc[4] = {};
  wave_gemm<192>(A + (size_t)m0 * 192, 192, W + (size_t)n0 * 192, 192, acc);
  const int lane = threadIdx.x & 31, nn = lane & 15, mh = lane >> 4;
#pragma unroll
  for (int t = 0; t < 4; ++t) {
    const int col = n0 + t * 16 + nn;
    const float bv = bias[col];
#pragma unroll
    for (int r = 0; r < 8; ++r) {
      const float u = acc[t][r] + bv;
      const float gl = 0.5f * u * (1.0f + erff(u * 0.70710678118654752f));
      out[(size_t)(m0 + mh * 8 + r) * 768 + col] = (_Float16)gl;
    }
  }
}

// fc2 GEMM + final residual -> d_out (f32)
__global__ void __launch_bounds__(256)
fc2_gemm(const _Float16* __restrict__ A, const _Float16* __restrict__ W,
         const float* __restrict__ bias, const float* __restrict__ x2,
         float* __restrict__ out) {
  const int wave = threadIdx.x >> 5;
  const int m0 = blockIdx.x * 128 + wave * 16;
  const int n0 = blockIdx.y * 64;
  v8f acc[4] = {};
  wave_gemm<768>(A + (size_t)m0 * 768, 768, W + (size_t)n0 * 768, 768, acc);
  const int lane = threadIdx.x & 31, nn = lane & 15, mh = lane >> 4;
#pragma unroll
  for (int t = 0; t < 4; ++t) {
    const int col = n0 + t * 16 + nn;
    const float bv = bias[col];
#pragma unroll
    for (int r = 0; r < 8; ++r) {
      const size_t idx = (size_t)(m0 + mh * 8 + r) * 192 + col;
      out[idx] = x2[idx] + acc[t][r] + bv;
    }
  }
}

// ---------------------------------------------------------------------------
// Attention: one block per (window, head). 128 threads = 4 waves.
// S = Q K^T (49 WMMAs, operands straight from global; K-dim=32 -> 1 step),
// softmax(+rel-pos bias +shift mask) in LDS, O = P V (zero-padded K=128).
// LDS ~62 KB: Vt (transposed V), Ss (f16 scores), Ps (f16 probs).
// ---------------------------------------------------------------------------
__global__ void __launch_bounds__(128)
attn_kernel(const _Float16* __restrict__ qkv, const float* __restrict__ biasb,
            const float* __restrict__ maskm, _Float16* __restrict__ outp) {
  __shared__ _Float16 Vt[32 * 128];    // V transposed: [d][token], pad tokens->128
  __shared__ _Float16 Ss[112 * 112];   // raw scores (f16)
  __shared__ _Float16 Ps[112 * 128];   // probabilities, K-padded to 128
  const int b_ = blockIdx.x, hh = blockIdx.y;
  const int tid = threadIdx.x;
  const _Float16* base = qkv + (size_t)b_ * 112 * 576;

  // stage V transposed into LDS (zero the K pad)
  for (int i = tid; i < 32 * 128; i += 128) {
    const int dd = i >> 7, n = i & 127;
    Vt[i] = (n < 112) ? base[(size_t)n * 576 + 384 + hh * 32 + dd] : (_Float16)0.0f;
  }
  __syncthreads();

  const int wave = tid >> 5;
  // S = Q K^T : 7x7 tiles of 16x16, single 16x16x32 WMMA each
  for (int t = wave; t < 49; t += 4) {
    const int mt = t / 7, nt = t - mt * 7;
    v16h a = load_a_frag(base + (size_t)(mt * 16) * 576 + hh * 32, 576);        // Q rows
    v16h b = load_b_frag(base + (size_t)(nt * 16) * 576 + 192 + hh * 32, 576);  // K rows
    v8f c = {};
    c = wmma_f16(a, b, c);
    const int lane = tid & 31, nn = lane & 15, mh = lane >> 4;
#pragma unroll
    for (int r = 0; r < 8; ++r)
      Ss[(mt * 16 + mh * 8 + r) * 112 + nt * 16 + nn] = (_Float16)c[r];
  }
  __syncthreads();

  // softmax per row (one thread per row)
  if (tid < 112) {
    _Float16* prow = Ps + (size_t)tid * 128;
    if (tid < 98) {
      _Float16* srow = Ss + (size_t)tid * 112;
      const float* bb = biasb + ((size_t)hh * 98 + tid) * 98;
      const float* mm = maskm + ((size_t)(b_ & 255) * 98 + tid) * 98;
      const float scale = 0.17677669529663687f;  // 1/sqrt(32)
      float mx = -3.0e38f;
      for (int m = 0; m < 98; ++m) {
        float l = (float)srow[m] * scale + bb[m] + mm[m];
        srow[m] = (_Float16)l;
        mx = fmaxf(mx, l);
      }
      float sum = 0.f;
      for (int m = 0; m < 98; ++m) {
        float e = __expf((float)srow[m] - mx);
        srow[m] = (_Float16)e;
        sum += e;
      }
      const float inv = 1.0f / sum;
      for (int m = 0; m < 98; ++m) prow[m] = (_Float16)((float)srow[m] * inv);
      for (int m = 98; m < 128; ++m) prow[m] = (_Float16)0.0f;
    } else {
      for (int m = 0; m < 128; ++m) prow[m] = (_Float16)0.0f;
    }
  }
  __syncthreads();

  // O = P V : 7x2 tiles, K = 128 (zero padded) -> 4 WMMA steps each
  for (int t = wave; t < 14; t += 4) {
    const int mt = t >> 1, nt = t & 1;
    v8f c = {};
#pragma unroll
    for (int ks = 0; ks < 128; ks += 32) {
      v16h a = load_a_frag(Ps + (size_t)(mt * 16) * 128 + ks, 128);
      v16h b = load_b_frag(Vt + (size_t)(nt * 16) * 128 + ks, 128);
      c = wmma_f16(a, b, c);
    }
    const int lane = tid & 31, nn = lane & 15, mh = lane >> 4;
#pragma unroll
    for (int r = 0; r < 8; ++r)
      outp[((size_t)b_ * 112 + mt * 16 + mh * 8 + r) * 192 + hh * 32 + nt * 16 + nn] =
          (_Float16)c[r];
  }
}

// ---------------------------------------------------------------------------
// host launcher
// ---------------------------------------------------------------------------
extern "C" void kernel_launch(void* const* d_in, const int* in_sizes, int n_in,
                              void* d_out, int out_size, void* d_ws, size_t ws_size,
                              hipStream_t stream) {
  (void)in_sizes; (void)n_in; (void)out_size; (void)ws_size;
  const float* x      = (const float*)d_in[0];
  const float* maskm  = (const float*)d_in[1];
  const float* n1g    = (const float*)d_in[2];
  const float* n1b    = (const float*)d_in[3];
  const float* qkv_w  = (const float*)d_in[4];
  const float* qkv_b  = (const float*)d_in[5];
  const float* rpb    = (const float*)d_in[6];
  const float* proj_w = (const float*)d_in[7];
  const float* proj_b = (const float*)d_in[8];
  const float* n2g    = (const float*)d_in[9];
  const float* n2b    = (const float*)d_in[10];
  const float* fc1_w  = (const float*)d_in[11];
  const float* fc1_b  = (const float*)d_in[12];
  const float* fc2_w  = (const float*)d_in[13];
  const float* fc2_b  = (const float*)d_in[14];
  float* out = (float*)d_out;

  char* ws = (char*)d_ws;
  size_t off = 0;
  auto take = [&](size_t bytes) -> char* {
    char* p = ws + off;
    off += (bytes + 255) & ~(size_t)255;
    return p;
  };
  _Float16* wXw  = (_Float16*)take((size_t)1024 * 112 * 192 * 2);  // partition; reused as attn out
  _Float16* wQKV = (_Float16*)take((size_t)1024 * 112 * 576 * 2);  // qkv; reused as h2 (LN2 out)
  float*    wX2  = (float*)   take((size_t)100352 * 192 * 4);      // shortcut+attn residual
  _Float16* wFC1 = (_Float16*)take((size_t)100352 * 768 * 2);      // gelu(fc1)
  _Float16* wQW  = (_Float16*)take((size_t)576 * 192 * 2);
  _Float16* wPW  = (_Float16*)take((size_t)192 * 192 * 2);
  _Float16* wF1W = (_Float16*)take((size_t)768 * 192 * 2);
  _Float16* wF2W = (_Float16*)take((size_t)192 * 768 * 2);
  float*    wBias= (float*)   take((size_t)6 * 9604 * 4);
  _Float16* wAttn = wXw;   // alias: Xw dead after qkv_gemm
  _Float16* wH2   = wQKV;  // alias: qkv dead after attention

  // weight conversions + bias gather
  cvt_f16_kernel<<<432, 256, 0, stream>>>(qkv_w, wQW, 110592);
  cvt_f16_kernel<<<144, 256, 0, stream>>>(proj_w, wPW, 36864);
  cvt_f16_kernel<<<576, 256, 0, stream>>>(fc1_w, wF1W, 147456);
  cvt_f16_kernel<<<576, 256, 0, stream>>>(fc2_w, wF2W, 147456);
  bias_gather_kernel<<<38, 256, 0, stream>>>(rpb, wBias);

  // LN1 + shift + window partition  (114688 padded tokens)
  ln1_shift_part<<<28672, 128, 0, stream>>>(x, n1g, n1b, wXw);

  // QKV: [114688 x 192] x [192 x 576]
  qkv_gemm<<<dim3(896, 9), 256, 0, stream>>>(wXw, wQW, qkv_b, wQKV);

  // windowed attention, 1024 windows x 6 heads
  attn_kernel<<<dim3(1024, 6), 128, 0, stream>>>(wQKV, wBias, maskm, wAttn);

  // proj + window-reverse + roll + residual
  proj_gemm<<<dim3(896, 3), 256, 0, stream>>>(wAttn, wPW, proj_b, x, wX2);

  // LN2 (100352 tokens)
  ln2_kernel<<<25088, 128, 0, stream>>>(wX2, n2g, n2b, wH2);

  // MLP
  fc1_gemm<<<dim3(784, 12), 256, 0, stream>>>(wH2, wF1W, fc1_b, wFC1);
  fc2_gemm<<<dim3(784, 3), 256, 0, stream>>>(wFC1, wF2W, fc2_b, wX2, out);
}
